// Attention_46643344835180
// MI455X (gfx1250) — matrix operand order repaired and checked by reading, not compiled, not run on hardware
//
#include <hip/hip_runtime.h>
#include <hip/hip_bf16.h>

typedef __attribute__((ext_vector_type(16))) _Float16 v16h;
typedef __attribute__((ext_vector_type(8)))  float    v8f;
typedef __attribute__((ext_vector_type(2)))  _Float16 half2_t;
typedef _Float16 half_t;

#define DM   768
#define NH   12
#define DH   64
#define SEQ  1024
#define BATCH 8

// ---------- fragment loaders (contiguous-k order k = 16*khalf + s, wave32) ----------

// A-matrix 16x32 f16: lane holds row m=lane&15; khalf=lane>>4 selects
// K set {0..15} (khalf=0) or {16..31} (khalf=1).
__device__ __forceinline__ int a_koff(int i, int khalf) {
    return khalf * 16 + i * 2;
}

__device__ __forceinline__ v16h load_afrag_f32(const float* __restrict__ row, int khalf) {
    v16h r;
#pragma unroll
    for (int i = 0; i < 8; ++i) {
        float2 f = *(const float2*)(row + a_koff(i, khalf));
        r[2 * i] = (_Float16)f.x;
        r[2 * i + 1] = (_Float16)f.y;
    }
    return r;
}

template <typename PT>
__device__ __forceinline__ v16h load_afrag_f16(PT row, int khalf) {
    v16h r;
#pragma unroll
    for (int i = 0; i < 8; ++i) {
        half2_t h = *(const half2_t*)(row + a_koff(i, khalf));
        r[2 * i] = h.x;
        r[2 * i + 1] = h.y;
    }
    return r;
}

// B-matrix 32x16 f16: lane holds col n=lane&15; half-wave gives K base
// (khalf*16). Caller passes pointer to the 16 contiguous K values.
__device__ __forceinline__ v16h load_bfrag_contig(const half_t* __restrict__ p) {
    v16h r;
#pragma unroll
    for (int i = 0; i < 8; ++i) {
        half2_t h = *(const half2_t*)(p + 2 * i);
        r[2 * i] = h.x;
        r[2 * i + 1] = h.y;
    }
    return r;
}

__device__ __forceinline__ v8f wmma_f16(v16h a, v16h b, v8f c) {
    return __builtin_amdgcn_wmma_f32_16x16x32_f16(false, a, false, b, (short)0, c, false, false);
}

// ---------- kernel 0: convert + transpose weights to f16 ----------
// WqT/WkT/WvT: [h][e=64][d=768] (K-contiguous for B-fragments)
// WoT:         [d=768][k=768]   where k = h*64+e
__global__ void cvt_weights_kernel(const float* __restrict__ WQ, const float* __restrict__ WK,
                                   const float* __restrict__ WV, const float* __restrict__ WO,
                                   half_t* __restrict__ WqT, half_t* __restrict__ WkT,
                                   half_t* __restrict__ WvT, half_t* __restrict__ WoT) {
    int idx = blockIdx.x * blockDim.x + threadIdx.x;
    const int N = NH * DM * DH;  // == 768*768
    if (idx >= N) return;
    int h = idx / (DM * DH);
    int r = idx % (DM * DH);
    int d = r / DH, e = r % DH;
    size_t dstT = (size_t)h * DH * DM + (size_t)e * DM + d;
    WqT[dstT] = (half_t)WQ[idx];
    WkT[dstT] = (half_t)WK[idx];
    WvT[dstT] = (half_t)WV[idx];
    int i2 = idx / DM, d2 = idx % DM;  // W_O flat [i2=h*64+e][d2]
    WoT[(size_t)d2 * DM + i2] = (half_t)WO[idx];
}

// ---------- kernel 1: QKV projection ----------
// grid (64, 36): y = t*12+h, t in {Q,K,V}. block 256 = 8 waves; wave tile 16x64.
__global__ void qkv_kernel(const float* __restrict__ x,
                           const half_t* __restrict__ WqT, const half_t* __restrict__ WkT,
                           const half_t* __restrict__ WvT,
                           const float* __restrict__ bQ, const float* __restrict__ bK,
                           const float* __restrict__ bV,
                           half_t* __restrict__ Qh, half_t* __restrict__ Kh,
                           half_t* __restrict__ Vt) {
    const int wave = threadIdx.x >> 5, lane = threadIdx.x & 31;
    const int t = blockIdx.y / NH, h = blockIdx.y % NH;
    const int m0 = blockIdx.x * 128 + wave * 16;
    const int khalf = lane >> 4, nl = lane & 15;

    const half_t* WT = (t == 0 ? WqT : (t == 1 ? WkT : WvT)) + (size_t)h * DH * DM;
    const float* bias = (t == 0 ? bQ : (t == 1 ? bK : bV)) + h * DH;

    v8f acc[4] = {v8f{}, v8f{}, v8f{}, v8f{}};
    const float* xrow = x + (size_t)(m0 + nl) * DM;

    for (int k0 = 0; k0 < DM; k0 += 32) {
        v16h a = load_afrag_f32(xrow + k0, khalf);
#pragma unroll
        for (int c = 0; c < 4; ++c) {
            v16h b = load_bfrag_contig(WT + (size_t)(c * 16 + nl) * DM + k0 + khalf * 16);
            acc[c] = wmma_f16(a, b, acc[c]);
        }
    }

    const int bb = m0 / SEQ;
    const int s0 = m0 % SEQ;
#pragma unroll
    for (int c = 0; c < 4; ++c) {
        int e = c * 16 + nl;
        float bv = bias[e];
#pragma unroll
        for (int i = 0; i < 8; ++i) {
            int srow = s0 + i + 8 * khalf;
            float val = acc[c][i] + bv;
            if (t == 2) {
                Vt[((size_t)(bb * NH + h) * DH + e) * SEQ + srow] = (half_t)val;
            } else {
                half_t* dst = (t == 0) ? Qh : Kh;
                dst[((size_t)(bb * NH + h) * SEQ + srow) * DH + e] = (half_t)val;
            }
        }
    }
}

// ---------- kernel 2: flash attention (causal, online softmax) ----------
// one wave = one (b,h,16-row q tile). block 256 = 8 independent waves.
__global__ void attn_kernel(const half_t* __restrict__ Qh, const half_t* __restrict__ Kh,
                            const half_t* __restrict__ Vt, half_t* __restrict__ Zh) {
    __shared__ half_t plds[8][16 * 32];
    const int wave = threadIdx.x >> 5, lane = threadIdx.x & 31;
    const int g = blockIdx.x * 8 + wave;
    const int qt = g & 63;
    const int bh = g >> 6;
    const int h = bh % NH, b = bh / NH;
    const int q0 = qt * 16;
    const int khalf = lane >> 4, nl = lane & 15;

    const half_t* Qbase = Qh + (size_t)(b * NH + h) * SEQ * DH;
    const half_t* Kbase = Kh + (size_t)(b * NH + h) * SEQ * DH;
    const half_t* Vbase = Vt + (size_t)(b * NH + h) * DH * SEQ;

    const v16h qa0 = load_afrag_f16(Qbase + (size_t)(q0 + nl) * DH, khalf);
    const v16h qa1 = load_afrag_f16(Qbase + (size_t)(q0 + nl) * DH + 32, khalf);

    v8f o[4] = {v8f{}, v8f{}, v8f{}, v8f{}};
    float mrun[8], lrun[8];
#pragma unroll
    for (int i = 0; i < 8; ++i) { mrun[i] = -__builtin_inff(); lrun[i] = 0.f; }

    for (int j0 = 0; j0 < q0 + 16; j0 += 32) {
        // scores: S0 = Q x K[j0..j0+15]^T, S1 = Q x K[j0+16..j0+31]^T
        v8f s0 = {}, s1 = {};
        {
            v16h bk = load_bfrag_contig(Kbase + (size_t)(j0 + nl) * DH + khalf * 16);
            s0 = wmma_f16(qa0, bk, s0);
            bk = load_bfrag_contig(Kbase + (size_t)(j0 + nl) * DH + 32 + khalf * 16);
            s0 = wmma_f16(qa1, bk, s0);
            bk = load_bfrag_contig(Kbase + (size_t)(j0 + 16 + nl) * DH + khalf * 16);
            s1 = wmma_f16(qa0, bk, s1);
            bk = load_bfrag_contig(Kbase + (size_t)(j0 + 16 + nl) * DH + 32 + khalf * 16);
            s1 = wmma_f16(qa1, bk, s1);
        }
        const int col0 = j0 + nl, col1 = col0 + 16;
#pragma unroll
        for (int i = 0; i < 8; ++i) {
            const int row = q0 + i + 8 * khalf;
            float v0 = s0[i] * 0.125f;  // 1/sqrt(64)
            float v1 = s1[i] * 0.125f;
            if (col0 > row) v0 = -__builtin_inff();
            if (col1 > row) v1 = -__builtin_inff();
            // row max over the 16 lanes of this half-wave (row is striped over cols)
            float loc = fmaxf(v0, v1);
#pragma unroll
            for (int msk = 1; msk < 16; msk <<= 1)
                loc = fmaxf(loc, __shfl_xor(loc, msk, 32));
            const float newm = fmaxf(mrun[i], loc);
            const float scale = __expf(mrun[i] - newm);
            const float p0 = __expf(v0 - newm);
            const float p1 = __expf(v1 - newm);
            float ps = p0 + p1;
#pragma unroll
            for (int msk = 1; msk < 16; msk <<= 1)
                ps += __shfl_xor(ps, msk, 32);
            lrun[i] = lrun[i] * scale + ps;
            mrun[i] = newm;
#pragma unroll
            for (int c = 0; c < 4; ++c) o[c][i] *= scale;
            // bounce P through LDS: C layout -> row-major 16x32 tile
            plds[wave][(i + 8 * khalf) * 32 + nl] = (half_t)p0;
            plds[wave][(i + 8 * khalf) * 32 + nl + 16] = (half_t)p1;
        }
        asm volatile("s_wait_dscnt 0" ::: "memory");  // per-wave LDS WAR/RAW fence
        const v16h pa = load_afrag_f16(&plds[wave][nl * 32], khalf);
#pragma unroll
        for (int c = 0; c < 4; ++c) {
            v16h bv = load_bfrag_contig(Vbase + (size_t)(c * 16 + nl) * SEQ + j0 + khalf * 16);
            o[c] = wmma_f16(pa, bv, o[c]);
        }
    }

    // epilogue: Z[b][row][h*64+e] = O/l  (f16, row-major for the final GEMM)
#pragma unroll
    for (int c = 0; c < 4; ++c) {
        const int e = c * 16 + nl;
#pragma unroll
        for (int i = 0; i < 8; ++i) {
            const int row = q0 + i + 8 * khalf;
            float val = o[c][i] / lrun[i];
            Zh[((size_t)b * SEQ + row) * DM + h * DH + e] = (half_t)val;
        }
    }
}

// ---------- kernel 3: output projection + b_O ----------
__global__ void oproj_kernel(const half_t* __restrict__ Zh, const half_t* __restrict__ WoT,
                             const float* __restrict__ bO, float* __restrict__ out) {
    const int wave = threadIdx.x >> 5, lane = threadIdx.x & 31;
    const int m0 = blockIdx.x * 128 + wave * 16;
    const int n0 = blockIdx.y * 64;
    const int khalf = lane >> 4, nl = lane & 15;

    v8f acc[4] = {v8f{}, v8f{}, v8f{}, v8f{}};
    const half_t* zrow = Zh + (size_t)(m0 + nl) * DM;

    for (int k0 = 0; k0 < DM; k0 += 32) {
        v16h a = load_afrag_f16(zrow + k0, khalf);
#pragma unroll
        for (int c = 0; c < 4; ++c) {
            v16h b = load_bfrag_contig(WoT + (size_t)(n0 + c * 16 + nl) * DM + k0 + khalf * 16);
            acc[c] = wmma_f16(a, b, acc[c]);
        }
    }
#pragma unroll
    for (int c = 0; c < 4; ++c) {
        const int col = n0 + c * 16 + nl;
        const float bo = bO[col];
#pragma unroll
        for (int i = 0; i < 8; ++i) {
            const int row = m0 + i + 8 * khalf;
            out[(size_t)row * DM + col] = acc[c][i] + bo;
        }
    }
}

extern "C" void kernel_launch(void* const* d_in, const int* in_sizes, int n_in,
                              void* d_out, int out_size, void* d_ws, size_t ws_size,
                              hipStream_t stream) {
    const float* x  = (const float*)d_in[0];
    const float* WQ = (const float*)d_in[1];
    const float* bQ = (const float*)d_in[2];
    const float* WK = (const float*)d_in[3];
    const float* bK = (const float*)d_in[4];
    const float* WV = (const float*)d_in[5];
    const float* bV = (const float*)d_in[6];
    const float* WO = (const float*)d_in[7];
    const float* bO = (const float*)d_in[8];
    float* out = (float*)d_out;

    // workspace layout (all sizes multiples of 256B)
    const size_t szW  = (size_t)NH * DH * DM * sizeof(half_t);       // 1,179,648
    const size_t szQ  = (size_t)BATCH * NH * SEQ * DH * sizeof(half_t); // 12,582,912
    const size_t szZ  = (size_t)BATCH * SEQ * DM * sizeof(half_t);   // 12,582,912
    char* ws = (char*)d_ws;
    half_t* WqT = (half_t*)(ws);
    half_t* WkT = (half_t*)(ws + szW);
    half_t* WvT = (half_t*)(ws + 2 * szW);
    half_t* WoT = (half_t*)(ws + 3 * szW);
    half_t* Qh  = (half_t*)(ws + 4 * szW);
    half_t* Kh  = (half_t*)(ws + 4 * szW + szQ);
    half_t* Vt  = (half_t*)(ws + 4 * szW + 2 * szQ);
    half_t* Zh  = (half_t*)(ws + 4 * szW + 3 * szQ);
    (void)szZ; (void)ws_size; (void)in_sizes; (void)n_in; (void)out_size;

    // 0: weight convert/transpose (768*768 = 589824 elements)
    cvt_weights_kernel<<<dim3((NH * DM * DH + 255) / 256), dim3(256), 0, stream>>>(
        WQ, WK, WV, WO, WqT, WkT, WvT, WoT);

    // 1: QKV projection. grid (8192/128, 3*12)
    qkv_kernel<<<dim3((BATCH * SEQ) / 128, 3 * NH), dim3(256), 0, stream>>>(
        x, WqT, WkT, WvT, bQ, bK, bV, Qh, Kh, Vt);

    // 2: flash attention. B*H*(S/16) waves / 8 per block
    attn_kernel<<<dim3(BATCH * NH * (SEQ / 16) / 8), dim3(256), 0, stream>>>(
        Qh, Kh, Vt, Zh);

    // 3: output projection. grid (8192/128, 768/64)
    oproj_kernel<<<dim3((BATCH * SEQ) / 128, DM / 64), dim3(256), 0, stream>>>(
        Zh, WoT, bO, out);
}
